// HRM_14027363189427
// MI455X (gfx1250) — matrix operand order repaired and checked by reading, not compiled, not run on hardware
//
#include <hip/hip_runtime.h>
#include <math.h>

// Problem constants
#define Bb 4
#define Ss 512
#define Vv 32000
#define Hh 1024
#define SKk 256
#define Tt 4
#define MM (Bb * Ss)       // 2048
#define H3 (3 * Hh)        // 3072
#define NBLK (Ss / Tt)     // 128

typedef __attribute__((ext_vector_type(16))) __bf16 v16bf;
typedef __attribute__((ext_vector_type(8)))  float  v8f;

__device__ __forceinline__ unsigned short f2bfu(float f) {
  union { float f; unsigned u; } v; v.f = f;
  return (unsigned short)((v.u + 0x7FFFu + ((v.u >> 16) & 1u)) >> 16);
}
__device__ __forceinline__ __bf16 u2bf(unsigned short s) {
  union { unsigned short s; __bf16 b; } o; o.s = s; return o.b;
}
// A-matrix (16x32 bf16) per-lane K offset for slot j (CDNA5 WMMA layout)
__device__ __forceinline__ int a_koff(int j, int lane) {
  return ((j < 8) ? j : j + 8) + ((lane >= 16) ? 8 : 0);
}
// B-matrix (32x16 bf16) per-lane K offset for slot j
__device__ __forceinline__ int b_koff(int j, int lane) {
  return ((j < 8) ? j : j + 8) + ((lane >= 16) ? 8 : 0);
}
__device__ __forceinline__ float sigf(float x) { return 1.0f / (1.0f + __expf(-x)); }

// --- CDNA5 async global->LDS copy (ASYNCcnt tracked, bypasses VGPRs) -------
__device__ __forceinline__ unsigned lds_off32(const void* p) {
  // LDS-aperture flat address truncates to the LDS byte offset (ISA 10.2)
  return (unsigned)(unsigned long long)p;
}
__device__ __forceinline__ void async_ld_b128(unsigned lds, const void* g) {
  asm volatile("global_load_async_to_lds_b128 %0, %1, off"
               :: "v"(lds), "v"(g) : "memory");
}
__device__ __forceinline__ void wait_async0() {
#if __has_builtin(__builtin_amdgcn_s_wait_asynccnt)
  __builtin_amdgcn_s_wait_asynccnt(0);
#else
  asm volatile("s_wait_asynccnt 0x0" ::: "memory");
#endif
}

// ---------------------------------------------------------------------------
// Weight pre-conversion: f32 -> bf16 (RNE), float4 -> packed uint2
// ---------------------------------------------------------------------------
__global__ void cvt_bf16(const float* __restrict__ src,
                         unsigned short* __restrict__ dst, long n4) {
  long q = (long)blockIdx.x * 256 + threadIdx.x;
  if (q >= n4) return;
  float4 f = ((const float4*)src)[q];
  unsigned lo = (unsigned)f2bfu(f.x) | ((unsigned)f2bfu(f.y) << 16);
  unsigned hi = (unsigned)f2bfu(f.z) | ((unsigned)f2bfu(f.w) << 16);
  ((uint2*)dst)[q] = make_uint2(lo, hi);
}

// ---------------------------------------------------------------------------
// Big GEMM: C[M][ldc] = A_bf16[M][K] @ Wb_bf16[N][ldw](cols co..co+K).T + bias
// 128x128 tile/WG, 8 waves (2x4), 4x2 WMMA subtiles per wave.
// Double-buffered LDS fed by global_load_async_to_lds_b128 (raw bf16).
// ---------------------------------------------------------------------------
__global__ __launch_bounds__(256) void gemm_bf16_wmma(
    const unsigned short* __restrict__ A, const unsigned short* __restrict__ W,
    const float* __restrict__ bias, float* __restrict__ C,
    int K, int ldw, int co, int ldc) {
  __shared__ unsigned short As[2][128][32];  // 16 KB
  __shared__ unsigned short Bs[2][128][32];  // 16 KB
  const int tid = threadIdx.x, lane = tid & 31, wave = tid >> 5;
  const int n0 = blockIdx.x * 128, m0 = blockIdx.y * 128;
  const int wm = wave >> 2, wn = wave & 3;
  const int nk = K >> 5;
  v8f acc[4][2] = {};

  auto stage = [&](int kc, int buf) {
    const int k0 = kc << 5;
#pragma unroll
    for (int i = 0; i < 2; ++i) {
      int hr = i * 256 + tid;          // 8-ushort chunk id over 128x4
      int m = hr >> 2, k8 = (hr & 3) * 8;
      async_ld_b128(lds_off32(&As[buf][m][k8]),
                    &A[(size_t)(m0 + m) * K + k0 + k8]);
      async_ld_b128(lds_off32(&Bs[buf][m][k8]),
                    &W[(size_t)(n0 + m) * ldw + co + k0 + k8]);
    }
  };

  stage(0, 0);
  for (int kc = 0; kc < nk; ++kc) {
    const int buf = kc & 1;
    wait_async0();        // my copies for chunk kc have landed
    __syncthreads();      // everyone's copies landed; prev compute done
    if (kc + 1 < nk) stage(kc + 1, buf ^ 1);  // overlap with compute below
    v16bf afr[4];
#pragma unroll
    for (int i = 0; i < 4; ++i)
#pragma unroll
      for (int j = 0; j < 16; ++j)
        afr[i][j] = u2bf(As[buf][wm * 64 + i * 16 + (lane & 15)][a_koff(j, lane)]);
#pragma unroll
    for (int ns = 0; ns < 2; ++ns) {
      v16bf bfr;
      int ncol = wn * 32 + ns * 16 + (lane & 15);
#pragma unroll
      for (int j = 0; j < 16; ++j) bfr[j] = u2bf(Bs[buf][ncol][b_koff(j, lane)]);
#pragma unroll
      for (int i = 0; i < 4; ++i)
        acc[i][ns] = __builtin_amdgcn_wmma_f32_16x16x32_bf16(
            false, afr[i], false, bfr, (short)0, acc[i][ns], false, false);
    }
  }
#pragma unroll
  for (int i = 0; i < 4; ++i)
#pragma unroll
    for (int ns = 0; ns < 2; ++ns) {
      int col = n0 + wn * 32 + ns * 16 + (lane & 15);
      float bv = bias[col];
#pragma unroll
      for (int v = 0; v < 8; ++v) {
        int row = m0 + wm * 64 + i * 16 + v + ((lane >= 16) ? 8 : 0);
        C[(size_t)row * ldc + col] = acc[i][ns][v] + bv;
      }
    }
}

// ---------------------------------------------------------------------------
// Gh_h = h_h[16x1024 bf16, rows>=4 zero] @ w_ih_l_bf16[:, 1024:2048].T -> [4][3072]
// grid 24 WGs x 128 cols; async bf16 W tile.
// ---------------------------------------------------------------------------
__global__ __launch_bounds__(256) void ghh_gemm(
    const unsigned short* __restrict__ hb, const unsigned short* __restrict__ W,
    float* __restrict__ GhH) {
  __shared__ unsigned short Ws[128][32];  // 8 KB
  const int tid = threadIdx.x, lane = tid & 31, wave = tid >> 5;
  const int n0 = blockIdx.x * 128;
  v8f acc = {};
  for (int k0 = 0; k0 < Hh; k0 += 32) {
#pragma unroll
    for (int i = 0; i < 2; ++i) {
      int q = i * 256 + tid;
      int n = q >> 2, k8 = (q & 3) * 8;
      async_ld_b128(lds_off32(&Ws[n][k8]),
                    &W[(size_t)(n0 + n) * (2 * Hh) + Hh + k0 + k8]);
    }
    wait_async0();
    __syncthreads();
    v16bf afr, bfr;
#pragma unroll
    for (int j = 0; j < 16; ++j)
      afr[j] = u2bf(hb[(lane & 15) * Hh + k0 + a_koff(j, lane)]);
    int ncol = wave * 16 + (lane & 15);
#pragma unroll
    for (int j = 0; j < 16; ++j) bfr[j] = u2bf(Ws[ncol][b_koff(j, lane)]);
    acc = __builtin_amdgcn_wmma_f32_16x16x32_bf16(false, afr, false, bfr,
                                                  (short)0, acc, false, false);
    __syncthreads();  // before next chunk overwrites Ws
  }
  if (lane < 16) {
    int col = n0 + wave * 16 + lane;
#pragma unroll
    for (int v = 0; v < 4; ++v) GhH[v * H3 + col] = acc[v];
  }
}

// ---------------------------------------------------------------------------
// One low-GRU step: gh = h_l @ w_hh_l.T (3 gates), gi = Gx[t] + Gh_h,
// h_new = (1-z)*n + z*h.  grid 8 WGs x 128 cols; async bf16 W tiles.
// ---------------------------------------------------------------------------
__global__ __launch_bounds__(256) void gru_low_step(
    const unsigned short* __restrict__ hcur, unsigned short* __restrict__ hnxt,
    float* __restrict__ hf, const unsigned short* __restrict__ Whh,
    const float* __restrict__ bhh, const float* __restrict__ Gx,
    const float* __restrict__ GhH, unsigned short* __restrict__ hs, int s) {
  __shared__ unsigned short Ws[3][128][32];  // 24 KB
  const int tid = threadIdx.x, lane = tid & 31, wave = tid >> 5;
  const int n0 = blockIdx.x * 128;
  v8f acc[3] = {};
  for (int k0 = 0; k0 < Hh; k0 += 32) {
#pragma unroll
    for (int i = 0; i < 6; ++i) {
      int q = i * 256 + tid;                 // 8-ushort chunk over 3*128*4
      int g = q >> 9, r = (q >> 2) & 127, k8 = (q & 3) * 8;
      async_ld_b128(lds_off32(&Ws[g][r][k8]),
                    &Whh[(size_t)(g * Hh + n0 + r) * Hh + k0 + k8]);
    }
    wait_async0();
    __syncthreads();
    v16bf afr;
#pragma unroll
    for (int j = 0; j < 16; ++j)
      afr[j] = u2bf(hcur[(lane & 15) * Hh + k0 + a_koff(j, lane)]);
    int ncol = wave * 16 + (lane & 15);
#pragma unroll
    for (int g = 0; g < 3; ++g) {
      v16bf bfr;
#pragma unroll
      for (int j = 0; j < 16; ++j) bfr[j] = u2bf(Ws[g][ncol][b_koff(j, lane)]);
      acc[g] = __builtin_amdgcn_wmma_f32_16x16x32_bf16(
          false, afr, false, bfr, (short)0, acc[g], false, false);
    }
    __syncthreads();
  }
  if (lane < 16) {
    int col = n0 + wave * 16 + lane;
#pragma unroll
    for (int v = 0; v < 4; ++v) {
      float ghr = acc[0][v] + bhh[col];
      float ghz = acc[1][v] + bhh[Hh + col];
      float ghn = acc[2][v] + bhh[2 * Hh + col];
      size_t grow = (size_t)(v * Ss + s) * H3;
      float gir = Gx[grow + col] + GhH[v * H3 + col];
      float giz = Gx[grow + Hh + col] + GhH[v * H3 + Hh + col];
      float gin = Gx[grow + 2 * Hh + col] + GhH[v * H3 + 2 * Hh + col];
      float r = sigf(gir + ghr);
      float z = sigf(giz + ghz);
      float nn = tanhf(gin + r * ghn);
      float ho = hf[v * Hh + col];
      float hv = (1.0f - z) * nn + z * ho;
      hf[v * Hh + col] = hv;
      unsigned short hbv = f2bfu(hv);
      hnxt[v * Hh + col] = hbv;
      hs[(size_t)(v * Ss + s) * Hh + col] = hbv;
    }
  }
}

// ---------------------------------------------------------------------------
// One high-GRU step: gi = h_l @ w_ih_h.T + b_ih_h ; gh = h_h @ w_hh_h.T + b_hh_h
// h_h_new = (1-z)*n + z*h_h + z_p.  grid 8 WGs x 128 cols; async bf16 tiles.
// ---------------------------------------------------------------------------
__global__ __launch_bounds__(256) void gru_high_step(
    const unsigned short* __restrict__ hl, const unsigned short* __restrict__ hhc,
    unsigned short* __restrict__ hhn, float* __restrict__ hhf,
    const unsigned short* __restrict__ Wih, const unsigned short* __restrict__ Whh,
    const float* __restrict__ bih, const float* __restrict__ bhh,
    const float* __restrict__ zp) {
  __shared__ unsigned short Wi[3][128][32];  // 24 KB
  __shared__ unsigned short Wh[3][128][32];  // 24 KB
  const int tid = threadIdx.x, lane = tid & 31, wave = tid >> 5;
  const int n0 = blockIdx.x * 128;
  v8f ai[3] = {}, ah[3] = {};
  for (int k0 = 0; k0 < Hh; k0 += 32) {
#pragma unroll
    for (int i = 0; i < 6; ++i) {
      int q = i * 256 + tid;                 // 8-ushort chunk over 3*128*4
      int g = q >> 9, r = (q >> 2) & 127, k8 = (q & 3) * 8;
      size_t off = (size_t)(g * Hh + n0 + r) * Hh + k0 + k8;
      async_ld_b128(lds_off32(&Wi[g][r][k8]), &Wih[off]);
      async_ld_b128(lds_off32(&Wh[g][r][k8]), &Whh[off]);
    }
    wait_async0();
    __syncthreads();
    v16bf afl, afh;
#pragma unroll
    for (int j = 0; j < 16; ++j) {
      int ko = k0 + a_koff(j, lane);
      afl[j] = u2bf(hl[(lane & 15) * Hh + ko]);
      afh[j] = u2bf(hhc[(lane & 15) * Hh + ko]);
    }
    int ncol = wave * 16 + (lane & 15);
#pragma unroll
    for (int g = 0; g < 3; ++g) {
      v16bf bi, bh;
#pragma unroll
      for (int j = 0; j < 16; ++j) {
        int ko = b_koff(j, lane);
        bi[j] = u2bf(Wi[g][ncol][ko]);
        bh[j] = u2bf(Wh[g][ncol][ko]);
      }
      ai[g] = __builtin_amdgcn_wmma_f32_16x16x32_bf16(false, afl, false, bi,
                                                      (short)0, ai[g], false, false);
      ah[g] = __builtin_amdgcn_wmma_f32_16x16x32_bf16(false, afh, false, bh,
                                                      (short)0, ah[g], false, false);
    }
    __syncthreads();
  }
  if (lane < 16) {
    int col = n0 + wave * 16 + lane;
#pragma unroll
    for (int v = 0; v < 4; ++v) {
      float gir = ai[0][v] + bih[col];
      float giz = ai[1][v] + bih[Hh + col];
      float gin = ai[2][v] + bih[2 * Hh + col];
      float ghr = ah[0][v] + bhh[col];
      float ghz = ah[1][v] + bhh[Hh + col];
      float ghn = ah[2][v] + bhh[2 * Hh + col];
      float r = sigf(gir + ghr);
      float z = sigf(giz + ghz);
      float nn = tanhf(gin + r * ghn);
      float ho = hhf[v * Hh + col];
      float hv = (1.0f - z) * nn + z * ho + zp[v * Hh + col];
      hhf[v * Hh + col] = hv;
      hhn[v * Hh + col] = f2bfu(hv);
    }
  }
}

// ---------------------------------------------------------------------------
// Small setup kernels
// ---------------------------------------------------------------------------
__global__ void gather_emb(const int* __restrict__ x, const float* __restrict__ emb,
                           unsigned short* __restrict__ xe) {
  // one float4 (4 elems) per thread
  size_t q = (size_t)blockIdx.x * 256 + threadIdx.x;  // over MM*Hh/4
  int m = (int)(q >> 8), k4 = (int)(q & 255) * 4;
  int tok = x[m];
  float4 f = *(const float4*)&emb[(size_t)tok * Hh + k4];
  unsigned lo = (unsigned)f2bfu(f.x) | ((unsigned)f2bfu(f.y) << 16);
  unsigned hi = (unsigned)f2bfu(f.z) | ((unsigned)f2bfu(f.w) << 16);
  *(uint2*)&xe[(size_t)m * Hh + k4] = make_uint2(lo, hi);
}

__global__ void zp_init(const float* __restrict__ zvec, const float* __restrict__ Wz,
                        const float* __restrict__ bz, float* __restrict__ zp,
                        float* __restrict__ hhf, unsigned short* __restrict__ hh0,
                        unsigned short* __restrict__ hh1) {
  int idx = blockIdx.x * 256 + threadIdx.x;  // over 16*Hh
  if (idx >= 16 * Hh) return;
  int m = idx >> 10, col = idx & 1023;
  if (m < Bb) {
    float a = bz[col];
    for (int k = 0; k < SKk; ++k) a += zvec[m * SKk + k] * Wz[(size_t)col * SKk + k];
    zp[idx] = a; hhf[idx] = a; hh0[idx] = f2bfu(a);
  } else {
    hh0[idx] = 0;
  }
  hh1[idx] = 0;
}

__global__ void init_hl(float* __restrict__ hlf, unsigned short* __restrict__ hl0,
                        unsigned short* __restrict__ hl1) {
  int idx = blockIdx.x * 256 + threadIdx.x;
  if (idx >= 16 * Hh) return;
  hl0[idx] = 0; hl1[idx] = 0;
  if (idx < Bb * Hh) hlf[idx] = 0.0f;
}

__global__ void write_states(const float* __restrict__ hlf,
                             const float* __restrict__ hhf, float* __restrict__ o) {
  int i = blockIdx.x * 256 + threadIdx.x;
  if (i < Bb * Hh) o[i] = hlf[i];
  else if (i < 2 * Bb * Hh) o[i] = hhf[i - Bb * Hh];
}

// ---------------------------------------------------------------------------
extern "C" void kernel_launch(void* const* d_in, const int* in_sizes, int n_in,
                              void* d_out, int out_size, void* d_ws, size_t ws_size,
                              hipStream_t stream) {
  const int*   x      = (const int*)d_in[0];
  const float* z_vec  = (const float*)d_in[1];
  const float* emb    = (const float*)d_in[2];
  const float* w_ih_l = (const float*)d_in[3];
  const float* w_hh_l = (const float*)d_in[4];
  const float* b_ih_l = (const float*)d_in[5];
  const float* b_hh_l = (const float*)d_in[6];
  const float* w_ih_h = (const float*)d_in[7];
  const float* w_hh_h = (const float*)d_in[8];
  const float* b_ih_h = (const float*)d_in[9];
  const float* b_hh_h = (const float*)d_in[10];
  const float* w_out  = (const float*)d_in[11];
  const float* b_out  = (const float*)d_in[12];
  const float* w_z    = (const float*)d_in[13];
  const float* b_z    = (const float*)d_in[14];
  float* out = (float*)d_out;

  char* p = (char*)d_ws;
  float*          Gx     = (float*)p;          p += (size_t)MM * H3 * 4;
  unsigned short* xe     = (unsigned short*)p; p += (size_t)MM * Hh * 2;
  unsigned short* hsb    = (unsigned short*)p; p += (size_t)MM * Hh * 2;
  float*          zp     = (float*)p;          p += (size_t)Bb * Hh * 4;
  float*          hlf    = (float*)p;          p += (size_t)Bb * Hh * 4;
  float*          hhf    = (float*)p;          p += (size_t)Bb * Hh * 4;
  float*          GhH    = (float*)p;          p += (size_t)Bb * H3 * 4;
  unsigned short* hlb[2];
  hlb[0] = (unsigned short*)p; p += 16 * Hh * 2;
  hlb[1] = (unsigned short*)p; p += 16 * Hh * 2;
  unsigned short* hhb[2];
  hhb[0] = (unsigned short*)p; p += 16 * Hh * 2;
  hhb[1] = (unsigned short*)p; p += 16 * Hh * 2;
  // bf16 weight shadows
  unsigned short* wihl_b = (unsigned short*)p; p += (size_t)H3 * 2 * Hh * 2;
  unsigned short* whhl_b = (unsigned short*)p; p += (size_t)H3 * Hh * 2;
  unsigned short* wihh_b = (unsigned short*)p; p += (size_t)H3 * Hh * 2;
  unsigned short* whhh_b = (unsigned short*)p; p += (size_t)H3 * Hh * 2;
  unsigned short* wout_b = (unsigned short*)p; p += (size_t)Vv * Hh * 2;

  // one-time f32 -> bf16 weight conversion (streaming pass, float4 granularity)
  {
    long n4;
    n4 = (long)H3 * 2 * Hh / 4;
    cvt_bf16<<<(unsigned)((n4 + 255) / 256), 256, 0, stream>>>(w_ih_l, wihl_b, n4);
    n4 = (long)H3 * Hh / 4;
    cvt_bf16<<<(unsigned)((n4 + 255) / 256), 256, 0, stream>>>(w_hh_l, whhl_b, n4);
    cvt_bf16<<<(unsigned)((n4 + 255) / 256), 256, 0, stream>>>(w_ih_h, wihh_b, n4);
    cvt_bf16<<<(unsigned)((n4 + 255) / 256), 256, 0, stream>>>(w_hh_h, whhh_b, n4);
    n4 = (long)Vv * Hh / 4;
    cvt_bf16<<<(unsigned)((n4 + 255) / 256), 256, 0, stream>>>(w_out, wout_b, n4);
  }

  // state init + z projection
  init_hl<<<64, 256, 0, stream>>>(hlf, hlb[0], hlb[1]);
  zp_init<<<64, 256, 0, stream>>>(z_vec, w_z, b_z, zp, hhf, hhb[0], hhb[1]);
  // embedding gather (fp32 -> bf16), float4 granularity
  gather_emb<<<(MM * Hh / 4) / 256, 256, 0, stream>>>(x, emb, xe);
  // Gx = x_emb @ w_ih_l[:, :H].T + b_ih_l   [2048 x 3072]
  gemm_bf16_wmma<<<dim3(H3 / 128, MM / 128), 256, 0, stream>>>(
      xe, wihl_b, b_ih_l, Gx, Hh, 2 * Hh, 0, H3);

  // sequential hierarchy: 128 blocks x (high step + Gh_h + 4 low steps)
  int cl = 0, ch = 0;
  for (int n = 0; n < NBLK; ++n) {
    if (n > 0) {
      gru_high_step<<<8, 256, 0, stream>>>(hlb[cl], hhb[ch], hhb[ch ^ 1], hhf,
                                           wihh_b, whhh_b, b_ih_h, b_hh_h, zp);
      ch ^= 1;
    }
    ghh_gemm<<<H3 / 128, 256, 0, stream>>>(hhb[ch], wihl_b, GhH);
    for (int t = 0; t < Tt; ++t) {
      int s = n * Tt + t;
      gru_low_step<<<8, 256, 0, stream>>>(hlb[cl], hlb[cl ^ 1], hlf, whhl_b,
                                          b_hh_l, Gx, GhH, hsb, s);
      cl ^= 1;
    }
  }

  // logits: out = hs @ w_out.T + b_out   [2048 x 32000]
  gemm_bf16_wmma<<<dim3(Vv / 128, MM / 128), 256, 0, stream>>>(
      hsb, wout_b, b_out, out, Hh, Hh, 0, Vv);
  // tail: h_l, h_h (fp32) appended after logits
  write_states<<<32, 256, 0, stream>>>(hlf, hhf, out + (size_t)MM * Vv);
}
